// ActionDecoder_8340826489616
// MI455X (gfx1250) — matrix operand order repaired and checked by reading, not compiled, not run on hardware
//
#include <hip/hip_runtime.h>

#define VD 512
#define HD 1024
#define BD 128
#define SD 256
#define TD 128

typedef __attribute__((ext_vector_type(16))) __bf16 v16bf;
typedef __attribute__((ext_vector_type(8)))  __bf16 v8bf;
typedef __attribute__((ext_vector_type(8)))  float  v8f;

__device__ __forceinline__ float sigmoidf_(float x){ return 1.0f/(1.0f+__expf(-x)); }

// ---------------- prep kernels ----------------
__global__ void k_convert_bf16(const float* __restrict__ in, __bf16* __restrict__ out, size_t n){
  size_t i = (size_t)blockIdx.x*blockDim.x + threadIdx.x;
  if (i < n) out[i] = (__bf16)in[i];
}

// xs[t][b][:] = emb[token], token = (t==0) ? SOS(0) : target[b][t-1]
__global__ void k_gather_embed(const int* __restrict__ tgt, const float* __restrict__ emb,
                               __bf16* __restrict__ xs){
  size_t i = (size_t)blockIdx.x*blockDim.x + threadIdx.x;
  if (i >= (size_t)TD*BD*HD) return;
  int h = (int)(i % HD);
  int b = (int)((i / HD) % BD);
  int t = (int)(i / ((size_t)HD*BD));
  int tok = (t == 0) ? 0 : tgt[b*TD + t - 1];
  xs[i] = (__bf16)emb[(size_t)tok*HD + h];
}

// Wcat[4H, 3H] = [W_ih | W_hh] (row-major, bf16)
__global__ void k_build_wcat(const float* __restrict__ wih, const float* __restrict__ whh,
                             __bf16* __restrict__ wcat){
  size_t i = (size_t)blockIdx.x*blockDim.x + threadIdx.x;
  const size_t n = (size_t)4*HD*3*HD;
  if (i >= n) return;
  int col = (int)(i % (3*HD));
  int row = (int)(i / (3*HD));
  float v = (col < 2*HD) ? wih[(size_t)row*2*HD + col] : whh[(size_t)row*HD + (col - 2*HD)];
  wcat[i] = (__bf16)v;
}

__global__ void k_add_bias(const float* __restrict__ a, const float* __restrict__ b,
                           float* __restrict__ out, int n){
  int i = blockIdx.x*blockDim.x + threadIdx.x;
  if (i < n) out[i] = a[i] + b[i];
}

__global__ void k_init_state(const float* __restrict__ h0, const float* __restrict__ c0,
                             float* __restrict__ h, __bf16* __restrict__ hb, float* __restrict__ c){
  int i = blockIdx.x*blockDim.x + threadIdx.x;
  if (i < BD*HD){ float v = h0[i]; h[i] = v; hb[i] = (__bf16)v; c[i] = c0[i]; }
}

// ---------------- WMMA GEMM: C[M,N] = A[M,K] @ B[N,K]^T + bias ----------------
// A,B bf16 row-major; C f32 or bf16. One 32x32 output tile per wave (2x2 WMMA
// tiles, each fragment reused twice -> 4 WMMAs per 8 b128 loads). K step 32.
// Fragment layouts per CDNA5 ISA 7.12.2 (wave32):
//   A 16x32: lanes 0-15 row M=l hold K={half*8+0..7, 16+half*8+0..7} as pairs.
//   B 32x16: lanes col N=l hold K={half*8+0..7, 16+half*8+0..7} (same as A).
//   C/D: vgpr r -> row = r + half*8, col = lane&15.
__device__ __forceinline__ v16bf load_afrag(const __bf16* __restrict__ row, int k0, int half){
  v8bf a0 = *(const v8bf*)(row + k0 + half*8);
  v8bf a1 = *(const v8bf*)(row + k0 + 16 + half*8);
  return __builtin_shufflevector(a0, a1, 0,1,2,3,4,5,6,7,8,9,10,11,12,13,14,15);
}

template<bool OUT_BF16>
__global__ void k_gemm_bf16(const __bf16* __restrict__ A, const __bf16* __restrict__ Bm,
                            const float* __restrict__ bias, void* __restrict__ Cp,
                            int M, int N, int K){
  int wave = (int)(threadIdx.x >> 5);
  int lane = (int)(threadIdx.x & 31);
  int tilesN = N >> 5;                             // 32-wide tiles
  int tile = (int)blockIdx.x * 8 + wave;           // wave-uniform
  if (tile >= (M >> 5) * tilesN) return;           // whole-wave exit -> EXEC all 1s at WMMA
  int tm = tile / tilesN, tn = tile - tm*tilesN;
  int half = lane >> 4, l = lane & 15;
  const __bf16* Arow0 = A  + ((size_t)(tm*32 + l))*K;
  const __bf16* Arow1 = Arow0 + (size_t)16*K;
  const __bf16* Brow0 = Bm + ((size_t)(tn*32 + l))*K;
  const __bf16* Brow1 = Brow0 + (size_t)16*K;
  v8f acc00 = {}, acc01 = {}, acc10 = {}, acc11 = {};
  for (int k0 = 0; k0 < K; k0 += 32){
    v16bf a0 = load_afrag(Arow0, k0, half);
    v16bf a1 = load_afrag(Arow1, k0, half);
    v16bf b0 = load_afrag(Brow0, k0, half);
    v16bf b1 = load_afrag(Brow1, k0, half);
    acc00 = __builtin_amdgcn_wmma_f32_16x16x32_bf16(false, a0, false, b0, (short)0, acc00, false, false);
    acc01 = __builtin_amdgcn_wmma_f32_16x16x32_bf16(false, a0, false, b1, (short)0, acc01, false, false);
    acc10 = __builtin_amdgcn_wmma_f32_16x16x32_bf16(false, a1, false, b0, (short)0, acc10, false, false);
    acc11 = __builtin_amdgcn_wmma_f32_16x16x32_bf16(false, a1, false, b1, (short)0, acc11, false, false);
  }
  int col0 = tn*32 + l, col1 = col0 + 16;
  float bv0 = bias ? bias[col0] : 0.0f;
  float bv1 = bias ? bias[col1] : 0.0f;
#pragma unroll
  for (int r = 0; r < 8; ++r){
    int row0 = tm*32 + r + half*8;
    int row1 = row0 + 16;
    float v00 = acc00[r] + bv0, v01 = acc01[r] + bv1;
    float v10 = acc10[r] + bv0, v11 = acc11[r] + bv1;
    if (OUT_BF16){
      __bf16* C = (__bf16*)Cp;
      C[(size_t)row0*N + col0] = (__bf16)v00;
      C[(size_t)row0*N + col1] = (__bf16)v01;
      C[(size_t)row1*N + col0] = (__bf16)v10;
      C[(size_t)row1*N + col1] = (__bf16)v11;
    } else {
      float* C = (float*)Cp;
      C[(size_t)row0*N + col0] = v00;
      C[(size_t)row0*N + col1] = v01;
      C[(size_t)row1*N + col0] = v10;
      C[(size_t)row1*N + col1] = v11;
    }
  }
}

// ---------------- attention: energies + softmax + context (one block per batch) ----------------
__global__ void k_attn(const float* __restrict__ q, const __bf16* __restrict__ keys,
                       const __bf16* __restrict__ enc, const float* __restrict__ Va,
                       const float* __restrict__ VaB, float* __restrict__ attn_out,
                       __bf16* __restrict__ xcat, int t){
  __shared__ float sm_q[HD];
  __shared__ float sm_va[HD];
  __shared__ float sm_red[SD];
  __shared__ float sm_w[SD];
  int b = blockIdx.x, tid = threadIdx.x;
#pragma unroll
  for (int j = 0; j < HD/SD; ++j){
    sm_q [tid + SD*j] = q[(size_t)b*HD + tid + SD*j];
    sm_va[tid + SD*j] = Va[tid + SD*j];
  }
  __syncthreads();
  // e[b,tid] = Va . tanh(q[b] + keys[b,tid]) + Va_b
  const __bf16* krow = keys + ((size_t)b*SD + tid)*HD;
  float e = VaB[0];
  for (int h0 = 0; h0 < HD; h0 += 8){
    v8bf kv = *(const v8bf*)(krow + h0);
#pragma unroll
    for (int j = 0; j < 8; ++j)
      e += sm_va[h0+j] * tanhf(sm_q[h0+j] + (float)kv[j]);
  }
  // softmax over S=256
  sm_red[tid] = e; __syncthreads();
  for (int off = SD/2; off > 0; off >>= 1){
    if (tid < off) sm_red[tid] = fmaxf(sm_red[tid], sm_red[tid+off]);
    __syncthreads();
  }
  float m = sm_red[0]; __syncthreads();
  float ex = __expf(e - m);
  sm_red[tid] = ex; __syncthreads();
  for (int off = SD/2; off > 0; off >>= 1){
    if (tid < off) sm_red[tid] += sm_red[tid+off];
    __syncthreads();
  }
  float w = ex / sm_red[0];
  sm_w[tid] = w;
  attn_out[((size_t)b*TD + t)*SD + tid] = w;
  __syncthreads();
  // ctx[b,:] = sum_s w[s] * enc[b,s,:]  -> write into xcat[:, H:2H] as bf16
  float acc0=0.f, acc1=0.f, acc2=0.f, acc3=0.f;
  const __bf16* erow = enc + (size_t)b*SD*HD;
  for (int s = 0; s < SD; ++s){
    float ws = sm_w[s];
    const __bf16* r = erow + (size_t)s*HD;
    acc0 += ws*(float)r[tid];
    acc1 += ws*(float)r[tid+SD];
    acc2 += ws*(float)r[tid+2*SD];
    acc3 += ws*(float)r[tid+3*SD];
  }
  __bf16* cdst = xcat + (size_t)b*3*HD + HD;
  cdst[tid]      = (__bf16)acc0;
  cdst[tid+SD]   = (__bf16)acc1;
  cdst[tid+2*SD] = (__bf16)acc2;
  cdst[tid+3*SD] = (__bf16)acc3;
}

// xcat[:,0:H] = emb_t ; xcat[:,2H:3H] = h (pre-update)
__global__ void k_pack(const __bf16* __restrict__ xs_t, const __bf16* __restrict__ hb,
                       __bf16* __restrict__ xcat){
  int i = blockIdx.x*blockDim.x + threadIdx.x;
  if (i >= BD*HD) return;
  int b = i / HD, h = i % HD;
  xcat[(size_t)b*3*HD + h]        = xs_t[i];
  xcat[(size_t)b*3*HD + 2*HD + h] = hb[i];
}

__global__ void k_lstm(const float* __restrict__ gates, float* __restrict__ c,
                       float* __restrict__ h, __bf16* __restrict__ hb){
  int i = blockIdx.x*blockDim.x + threadIdx.x;
  if (i >= BD*HD) return;
  int b = i / HD, hh = i % HD;
  const float* g = gates + (size_t)b*4*HD;
  float gi = sigmoidf_(g[hh]);
  float gf = sigmoidf_(g[HD + hh]);
  float gg = tanhf    (g[2*HD + hh]);
  float go = sigmoidf_(g[3*HD + hh]);
  float cn = gf*c[i] + gi*gg;
  c[i] = cn;
  float hn = go*tanhf(cn);
  h[i] = hn; hb[i] = (__bf16)hn;
}

__global__ void k_logsoftmax(const float* __restrict__ logits, float* __restrict__ out, int t){
  __shared__ float sm[256];
  int b = blockIdx.x, tid = threadIdx.x;
  float x0 = logits[(size_t)b*VD + tid];
  float x1 = logits[(size_t)b*VD + tid + 256];
  sm[tid] = fmaxf(x0, x1); __syncthreads();
  for (int off = 128; off > 0; off >>= 1){
    if (tid < off) sm[tid] = fmaxf(sm[tid], sm[tid+off]);
    __syncthreads();
  }
  float m = sm[0]; __syncthreads();
  sm[tid] = __expf(x0-m) + __expf(x1-m); __syncthreads();
  for (int off = 128; off > 0; off >>= 1){
    if (tid < off) sm[tid] += sm[tid+off];
    __syncthreads();
  }
  float ls = __logf(sm[0]);
  float* dst = out + ((size_t)b*TD + t)*VD;
  dst[tid]     = x0 - m - ls;
  dst[tid+256] = x1 - m - ls;
}

// ---------------- host ----------------
extern "C" void kernel_launch(void* const* d_in, const int* in_sizes, int n_in,
                              void* d_out, int out_size, void* d_ws, size_t ws_size,
                              hipStream_t stream) {
  (void)in_sizes; (void)n_in; (void)out_size; (void)ws_size;
  const float* enc_f32 = (const float*)d_in[0];
  const float* h0      = (const float*)d_in[1];
  const float* c0      = (const float*)d_in[2];
  const int*   tgt     = (const int*)  d_in[3];
  /* d_in[4] = max_length (fixed = TD) */
  const float* emb     = (const float*)d_in[5];
  const float* Wa_w    = (const float*)d_in[6];
  const float* Wa_b    = (const float*)d_in[7];
  const float* Ua_w    = (const float*)d_in[8];
  const float* Ua_b    = (const float*)d_in[9];
  const float* Va_w    = (const float*)d_in[10];
  const float* Va_b    = (const float*)d_in[11];
  const float* W_ih    = (const float*)d_in[12];
  const float* W_hh    = (const float*)d_in[13];
  const float* b_ih    = (const float*)d_in[14];
  const float* b_hh    = (const float*)d_in[15];
  const float* out_w   = (const float*)d_in[16];
  const float* out_b   = (const float*)d_in[17];

  char* ws = (char*)d_ws;
  size_t off = 0;
  auto carve = [&](size_t bytes)->char*{
    char* p = ws + off;
    off += (bytes + 255) & ~(size_t)255;
    return p;
  };
  __bf16* enc_bf  = (__bf16*)carve((size_t)BD*SD*HD*2);
  __bf16* keys_bf = (__bf16*)carve((size_t)BD*SD*HD*2);
  __bf16* xs_bf   = (__bf16*)carve((size_t)TD*BD*HD*2);
  __bf16* Wa_bf   = (__bf16*)carve((size_t)HD*HD*2);
  __bf16* Ua_bf   = (__bf16*)carve((size_t)HD*HD*2);
  __bf16* Wcat_bf = (__bf16*)carve((size_t)4*HD*3*HD*2);
  __bf16* ow_bf   = (__bf16*)carve((size_t)VD*HD*2);
  float*  bias_c  = (float*) carve((size_t)4*HD*4);
  float*  qbuf    = (float*) carve((size_t)BD*HD*4);
  __bf16* xcat    = (__bf16*)carve((size_t)BD*3*HD*2);
  float*  gatesb  = (float*) carve((size_t)BD*4*HD*4);
  float*  hbuf    = (float*) carve((size_t)BD*HD*4);
  __bf16* hbf     = (__bf16*)carve((size_t)BD*HD*2);
  float*  cbuf    = (float*) carve((size_t)BD*HD*4);
  float*  logitsb = (float*) carve((size_t)BD*VD*4);

  const int TPB = 256;
  auto nb = [](size_t n){ return (unsigned)((n + 255)/256); };

  // One-time prep (runs every call; deterministic)
  k_convert_bf16<<<nb((size_t)BD*SD*HD), TPB, 0, stream>>>(enc_f32, enc_bf, (size_t)BD*SD*HD);
  k_convert_bf16<<<nb((size_t)HD*HD),    TPB, 0, stream>>>(Wa_w,  Wa_bf, (size_t)HD*HD);
  k_convert_bf16<<<nb((size_t)HD*HD),    TPB, 0, stream>>>(Ua_w,  Ua_bf, (size_t)HD*HD);
  k_convert_bf16<<<nb((size_t)VD*HD),    TPB, 0, stream>>>(out_w, ow_bf, (size_t)VD*HD);
  k_gather_embed<<<nb((size_t)TD*BD*HD), TPB, 0, stream>>>(tgt, emb, xs_bf);
  k_build_wcat<<<nb((size_t)4*HD*3*HD),  TPB, 0, stream>>>(W_ih, W_hh, Wcat_bf);
  k_add_bias<<<nb(4*HD), TPB, 0, stream>>>(b_ih, b_hh, bias_c, 4*HD);
  k_init_state<<<nb(BD*HD), TPB, 0, stream>>>(h0, c0, hbuf, hbf, cbuf);

  // keys_proj = enc @ Ua^T + Ua_b  (loop-invariant, stored bf16 so it lives in L2)
  {
    int M = BD*SD, N = HD, K = HD;
    int tiles = (M/32)*(N/32);
    k_gemm_bf16<true><<<(tiles+7)/8, 256, 0, stream>>>(enc_bf, Ua_bf, Ua_b, keys_bf, M, N, K);
  }

  float* out_logprob = (float*)d_out;                           // [B,T,V]
  float* out_hidden  = out_logprob + (size_t)BD*TD*VD;          // [1,B,H]
  float* out_attn    = out_hidden  + (size_t)BD*HD;             // [B,T,S]

  for (int t = 0; t < TD; ++t){
    { int M=BD, N=HD, K=HD; int tiles=(M/32)*(N/32);            // q = h @ Wa^T + Wa_b
      k_gemm_bf16<false><<<(tiles+7)/8, 256, 0, stream>>>(hbf, Wa_bf, Wa_b, qbuf, M, N, K); }
    k_attn<<<BD, SD, 0, stream>>>(qbuf, keys_bf, enc_bf, Va_w, Va_b, out_attn, xcat, t);
    k_pack<<<nb(BD*HD), TPB, 0, stream>>>(xs_bf + (size_t)t*BD*HD, hbf, xcat);
    { int M=BD, N=4*HD, K=3*HD; int tiles=(M/32)*(N/32);        // gates = [x,ctx,h] @ Wcat^T + b
      k_gemm_bf16<false><<<(tiles+7)/8, 256, 0, stream>>>(xcat, Wcat_bf, bias_c, gatesb, M, N, K); }
    k_lstm<<<nb(BD*HD), TPB, 0, stream>>>(gatesb, cbuf, hbuf, hbf);
    { int M=BD, N=VD, K=HD; int tiles=(M/32)*(N/32);            // logits = h_new @ out_w^T + out_b
      k_gemm_bf16<false><<<(tiles+7)/8, 256, 0, stream>>>(hbf, ow_bf, out_b, logitsb, M, N, K); }
    k_logsoftmax<<<BD, 256, 0, stream>>>(logitsb, out_logprob, t);
  }
  hipMemcpyAsync(out_hidden, hbuf, (size_t)BD*HD*sizeof(float), hipMemcpyDeviceToDevice, stream);
}
